// TodTransformerEncoderLayer_32615981646425
// MI455X (gfx1250) — compile-verified, matrix operand order checked by reading, not run on hardware
//
#include <hip/hip_runtime.h>
#include <hip/hip_bf16.h>

// ---------------------------------------------------------------------------
// Deformable-DETR encoder layer for MI455X (gfx1250, wave32, WMMA).
// B=2, LQ=10723, D=256, NH=8, HD=32, NL=4, NP=4, DFF=1024.
// GEMMs: v_wmma_f32_16x16x32_bf16, fp32 accumulation. Weights pre-transposed
// to N x K bf16; the 64xK weight tile is staged per block into LDS with
// double-buffered global_load_async_to_lds_b128 (ASYNCcnt), A streams from
// HBM with global_load_b128 + prefetch.
// ---------------------------------------------------------------------------

typedef __attribute__((ext_vector_type(16))) __bf16 v16bf;
typedef __attribute__((ext_vector_type(8)))  float  v8f;

#define D_MODEL 256
#define N_HEADS 8
#define H_DIM   32
#define N_LVL   4
#define N_PT    4
#define BROW    40   // padded LDS row stride (shorts) for a 32-wide k-chunk

__device__ __forceinline__ unsigned short f32_to_bf16(float f) {
    unsigned int u = __builtin_bit_cast(unsigned int, f);
    unsigned int r = u + 0x7FFFu + ((u >> 16) & 1u);   // round-to-nearest-even
    return (unsigned short)(r >> 16);
}

struct Frag32B { uint4 lo, hi; };

// 16x32 bf16 WMMA fragment chunk for this lane (ISA 16-bit A/B layout):
// lane<16 -> K[kb..kb+7] and K[kb+16..kb+23], kb = (lane>>4)*8.
__device__ __forceinline__ v16bf load_frag(const unsigned short* p) {
    Frag32B f;
    f.lo = *reinterpret_cast<const uint4*>(p);
    f.hi = *reinterpret_cast<const uint4*>(p + 16);
    return __builtin_bit_cast(v16bf, f);
}

// LDS-sourced fragment: two 16B chunks 32B apart (padded row).
__device__ __forceinline__ v16bf load_frag_lds(const unsigned short* p) {
    Frag32B f;
    f.lo = *reinterpret_cast<const uint4*>(p);
    f.hi = *reinterpret_cast<const uint4*>(p + 16);
    return __builtin_bit_cast(v16bf, f);
}

// Async copy 16B global -> LDS (ASYNCcnt-tracked).
__device__ __forceinline__ void async_copy_b128(unsigned lds_off, const void* gaddr) {
    asm volatile("global_load_async_to_lds_b128 %0, %1, off"
                 :: "v"(lds_off), "v"(gaddr) : "memory");
}
__device__ __forceinline__ void wait_async0() {
    asm volatile("s_wait_asynccnt 0x0" ::: "memory");
}
__device__ __forceinline__ void wait_async1() {
    asm volatile("s_wait_asynccnt 0x1" ::: "memory");
}

// ---------------------------------------------------------------------------
// 1) prep: q_bf16 = bf16(query + query_pos); value_bf16 = bf16(value)
// ---------------------------------------------------------------------------
__global__ void prep_kernel(const float* __restrict__ q,
                            const float* __restrict__ qp,
                            const float* __restrict__ val,
                            unsigned short* __restrict__ qb,
                            unsigned short* __restrict__ vb,
                            int n) {
    int i = blockIdx.x * blockDim.x + threadIdx.x;
    if (i < n) {
        qb[i] = f32_to_bf16(q[i] + qp[i]);
        vb[i] = f32_to_bf16(val[i]);
    }
}

// ---------------------------------------------------------------------------
// 2) weight transpose: f32 (K x N) -> bf16 (N x K)
// ---------------------------------------------------------------------------
__global__ void transpose_w_kernel(const float* __restrict__ in,
                                   unsigned short* __restrict__ out,
                                   int K, int N) {
    int i = blockIdx.x * blockDim.x + threadIdx.x;
    if (i < K * N) {
        int k = i / N, n = i % N;
        out[(size_t)n * K + k] = f32_to_bf16(in[i]);
    }
}

// ---------------------------------------------------------------------------
// 3) WMMA GEMM: C[M,N] = A[M,K](bf16) @ Wt[N,K]^T(bf16) + bias, opt. ReLU.
//    Block = 8 waves; block tile 128(M) x 64(N); each wave owns 16x64.
//    Weight k-chunks (64x32) staged to LDS via double-buffered async copies.
// ---------------------------------------------------------------------------
template <bool RELU, bool OUT_BF16>
__global__ __launch_bounds__(256) void gemm_bf16_wmma_kernel(
    const unsigned short* __restrict__ A,   // M x K
    const unsigned short* __restrict__ Wt,  // N x K
    const float* __restrict__ bias,         // N
    float* __restrict__ Cf,
    unsigned short* __restrict__ Cb,
    int M, int N, int K) {
    __shared__ __align__(16) unsigned short sB[2][64 * BROW];

    const int tid    = threadIdx.x;
    const int lane   = tid & 31;
    const int wave   = tid >> 5;
    const int l15    = lane & 15;
    const int lhalf  = lane >> 4;
    const int m_wave = blockIdx.x * 128 + wave * 16;
    const int n_base = blockIdx.y * 64;

    int arow = m_wave + l15;
    if (arow >= M) arow = M - 1;                 // clamp: OOB rows never stored
    const unsigned short* Ap = A + (size_t)arow * K + lhalf * 8;

    // Per-thread share of the 64x32 weight-slice copy (256 thr * 16B = 4KB).
    const int cr = tid >> 2;            // row 0..63 of the N-tile
    const int cc = (tid & 3) * 8;       // k-chunk column 0,8,16,24
    const unsigned short* wsrc = Wt + (size_t)(n_base + cr) * K + cc;
    const unsigned lds_base = (unsigned)(uintptr_t)(&sB[0][0]);
    const unsigned lds_dst  = lds_base + (unsigned)(cr * BROW + cc) * 2u;
    const unsigned buf_bytes = 64u * BROW * 2u;

    // prologue: stage chunk 0 into buffer 0
    async_copy_b128(lds_dst, wsrc);

    v8f acc[4] = {};
    const int nk = K >> 5;
    for (int i = 0; i < nk; ++i) {
        if (i + 1 < nk) {  // stage next chunk into the other buffer
            async_copy_b128(lds_dst + ((unsigned)(i + 1) & 1u) * buf_bytes,
                            wsrc + (i + 1) * 32);
            wait_async1();                 // chunk i resident (in-order)
        } else {
            wait_async0();
        }
        __syncthreads();

        v16bf a = load_frag(Ap + i * 32);
        if (i + 1 < nk) __builtin_prefetch(Ap + (i + 1) * 32, 0, 0);
        const unsigned short* btile = &sB[i & 1][0];
#pragma unroll
        for (int j = 0; j < 4; ++j) {
            v16bf b = load_frag_lds(btile + (j * 16 + l15) * BROW + lhalf * 8);
            acc[j] = __builtin_amdgcn_wmma_f32_16x16x32_bf16(
                false, a, false, b, (short)0, acc[j], false, false);
        }
        __syncthreads();                   // reads done before next overwrite
    }

#pragma unroll
    for (int j = 0; j < 4; ++j) {
        const int col = n_base + j * 16 + l15;
        const float bv = bias[col];
#pragma unroll
        for (int r = 0; r < 8; ++r) {
            const int row = m_wave + lhalf * 8 + r;   // C layout: VGPR r -> M
            if (row < M) {
                float y = acc[j][r] + bv;
                if (RELU) y = fmaxf(y, 0.0f);
                if (OUT_BF16) Cb[(size_t)row * N + col] = f32_to_bf16(y);
                else          Cf[(size_t)row * N + col] = y;
            }
        }
    }
}

// ---------------------------------------------------------------------------
// 4) softmax over 16 contiguous logits per (b,q,h)  (in place)
// ---------------------------------------------------------------------------
__global__ void softmax16_kernel(float* __restrict__ attn, int groups) {
    int g = blockIdx.x * blockDim.x + threadIdx.x;
    if (g >= groups) return;
    float4* p = reinterpret_cast<float4*>(attn + (size_t)g * 16);
    float4 v[4] = {p[0], p[1], p[2], p[3]};
    float* s = reinterpret_cast<float*>(v);
    float m = s[0];
#pragma unroll
    for (int i = 1; i < 16; ++i) m = fmaxf(m, s[i]);
    float sum = 0.f;
#pragma unroll
    for (int i = 0; i < 16; ++i) { s[i] = __expf(s[i] - m); sum += s[i]; }
    float inv = 1.0f / sum;
#pragma unroll
    for (int i = 0; i < 16; ++i) s[i] *= inv;
    p[0] = v[0]; p[1] = v[1]; p[2] = v[2]; p[3] = v[3];
}

// ---------------------------------------------------------------------------
// 5) MS-deformable sampling: one wave per (b,q,h); lane = head-dim channel.
//    x = ref_x*W + off_x - 0.5 (offset normalizer folds out analytically).
// ---------------------------------------------------------------------------
__global__ __launch_bounds__(256) void msda_sample_kernel(
    const float* __restrict__ vproj,   // (B, Lq, 256)
    const float* __restrict__ off,     // (B*Lq, 256)  [h*32 + p*2 + c]
    const float* __restrict__ attn,    // (B*Lq, 128)  [h*16 + p]
    const float* __restrict__ ref,     // (B*Lq, 8)    [lev*2 + c]
    unsigned short* __restrict__ out_bf16,  // (B*Lq, 256)
    int Bn, int Lq) {
    const int HH[N_LVL] = {76, 38, 19, 10};
    const int WW[N_LVL] = {106, 53, 27, 14};
    const int LS[N_LVL] = {0, 8056, 10070, 10583};

    const int lane = threadIdx.x & 31;
    const int wid  = (blockIdx.x * blockDim.x + threadIdx.x) >> 5;
    const int total = Bn * Lq * N_HEADS;
    if (wid >= total) return;
    const int h  = wid % N_HEADS;
    const int bq = wid / N_HEADS;
    const int b  = bq / Lq;

    float a_l = 0.f, ox_l = 0.f, oy_l = 0.f, ref_l = 0.f;
    if (lane < 16) {
        a_l  = attn[(size_t)bq * 128 + h * 16 + lane];
        ox_l = off[(size_t)bq * 256 + h * 32 + lane * 2 + 0];
        oy_l = off[(size_t)bq * 256 + h * 32 + lane * 2 + 1];
    }
    if (lane < 8) ref_l = ref[(size_t)bq * 8 + lane];

    float acc = 0.f;
#pragma unroll
    for (int p = 0; p < 16; ++p) {
        const int lev = p >> 2;
        const float w  = __shfl(a_l, p, 32);
        const float ox = __shfl(ox_l, p, 32);
        const float oy = __shfl(oy_l, p, 32);
        const float rx = __shfl(ref_l, lev * 2 + 0, 32);
        const float ry = __shfl(ref_l, lev * 2 + 1, 32);
        const int Wl = WW[lev], Hl = HH[lev];
        const float x = rx * (float)Wl + ox - 0.5f;
        const float y = ry * (float)Hl + oy - 0.5f;
        const float x0f = floorf(x), y0f = floorf(y);
        const float fx = x - x0f, fy = y - y0f;
        const int x0 = (int)x0f, y0 = (int)y0f;
        const float* vbase =
            vproj + ((size_t)b * Lq + LS[lev]) * D_MODEL + h * H_DIM + lane;
#pragma unroll
        for (int dy = 0; dy < 2; ++dy) {
#pragma unroll
            for (int dx = 0; dx < 2; ++dx) {
                const int xi = x0 + dx, yi = y0 + dy;
                const bool valid = (xi >= 0) & (xi < Wl) & (yi >= 0) & (yi < Hl);
                const int xc = xi < 0 ? 0 : (xi >= Wl ? Wl - 1 : xi);
                const int yc = yi < 0 ? 0 : (yi >= Hl ? Hl - 1 : yi);
                const float wc = (dx ? fx : 1.0f - fx) * (dy ? fy : 1.0f - fy);
                const float v = vbase[(size_t)(yc * Wl + xc) * D_MODEL];
                acc += valid ? v * (wc * w) : 0.0f;
            }
        }
    }
    out_bf16[(size_t)bq * D_MODEL + h * H_DIM + lane] = f32_to_bf16(acc);
}

// ---------------------------------------------------------------------------
// 6) fused residual add + LayerNorm; one wave per 256-wide row.
// ---------------------------------------------------------------------------
template <bool WRITE_BF16>
__global__ __launch_bounds__(256) void add_ln_kernel(
    const float* __restrict__ x, const float* __restrict__ r,
    const float* __restrict__ g, const float* __restrict__ bt,
    float* __restrict__ outf, unsigned short* __restrict__ outb, int rows) {
    const int lane = threadIdx.x & 31;
    const int row  = (blockIdx.x * blockDim.x + threadIdx.x) >> 5;
    if (row >= rows) return;
    const float* px = x + (size_t)row * D_MODEL;
    const float* pr = r + (size_t)row * D_MODEL;
    float v[8], s = 0.f, s2 = 0.f;
#pragma unroll
    for (int i = 0; i < 8; ++i) {
        const int c = lane + 32 * i;
        const float t = px[c] + pr[c];
        v[i] = t; s += t; s2 += t * t;
    }
#pragma unroll
    for (int o = 16; o > 0; o >>= 1) {
        s  += __shfl_xor(s, o, 32);
        s2 += __shfl_xor(s2, o, 32);
    }
    const float mu   = s * (1.0f / D_MODEL);
    const float var  = s2 * (1.0f / D_MODEL) - mu * mu;
    const float rstd = rsqrtf(var + 1e-5f);
#pragma unroll
    for (int i = 0; i < 8; ++i) {
        const int c = lane + 32 * i;
        const float y = (v[i] - mu) * rstd * g[c] + bt[c];
        outf[(size_t)row * D_MODEL + c] = y;
        if (WRITE_BF16) outb[(size_t)row * D_MODEL + c] = f32_to_bf16(y);
    }
}

// ---------------------------------------------------------------------------
// launcher
// ---------------------------------------------------------------------------
extern "C" void kernel_launch(void* const* d_in, const int* in_sizes, int n_in,
                              void* d_out, int out_size, void* d_ws, size_t ws_size,
                              hipStream_t stream) {
    (void)in_sizes; (void)n_in; (void)out_size; (void)ws_size;
    const int Bn = 2, LQ = 10723, Dm = D_MODEL, Dff = 1024;
    const int M = Bn * LQ;                      // 21446 rows

    const float* query = (const float*)d_in[0];
    const float* qpos  = (const float*)d_in[1];
    const float* value = (const float*)d_in[2];
    const float* refp  = (const float*)d_in[3];
    const float* W_off = (const float*)d_in[6];
    const float* b_off = (const float*)d_in[7];
    const float* W_att = (const float*)d_in[8];
    const float* b_att = (const float*)d_in[9];
    const float* W_v   = (const float*)d_in[10];
    const float* b_v   = (const float*)d_in[11];
    const float* W_out = (const float*)d_in[12];
    const float* b_out = (const float*)d_in[13];
    const float* ln1g  = (const float*)d_in[14];
    const float* ln1b  = (const float*)d_in[15];
    const float* W1    = (const float*)d_in[16];
    const float* b1    = (const float*)d_in[17];
    const float* W2    = (const float*)d_in[18];
    const float* b2    = (const float*)d_in[19];
    const float* ln2g  = (const float*)d_in[20];
    const float* ln2b  = (const float*)d_in[21];
    float* out = (float*)d_out;

    char* wsp = (char*)d_ws;
    auto alloc = [&](size_t bytes) -> char* {
        char* p = wsp;
        wsp += (bytes + 255) & ~(size_t)255;
        return p;
    };
    unsigned short* q_bf   = (unsigned short*)alloc((size_t)M * Dm * 2);
    unsigned short* v_bf   = (unsigned short*)alloc((size_t)M * Dm * 2);
    unsigned short* Wt_v   = (unsigned short*)alloc((size_t)Dm * Dm * 2);
    unsigned short* Wt_off = (unsigned short*)alloc((size_t)Dm * Dm * 2);
    unsigned short* Wt_att = (unsigned short*)alloc((size_t)Dm * 128 * 2);
    unsigned short* Wt_out = (unsigned short*)alloc((size_t)Dm * Dm * 2);
    unsigned short* Wt_1   = (unsigned short*)alloc((size_t)Dm * Dff * 2);
    unsigned short* Wt_2   = (unsigned short*)alloc((size_t)Dff * Dm * 2);
    float* v_proj          = (float*)alloc((size_t)M * Dm * 4);
    float* off_buf         = (float*)alloc((size_t)M * Dm * 4);
    float* attn_buf        = (float*)alloc((size_t)M * 128 * 4);
    unsigned short* samp   = (unsigned short*)alloc((size_t)M * Dm * 2);
    float* src2            = (float*)alloc((size_t)M * Dm * 4);
    float* q1              = (float*)alloc((size_t)M * Dm * 4);
    unsigned short* q1_bf  = (unsigned short*)alloc((size_t)M * Dm * 2);
    unsigned short* ffn_h  = (unsigned short*)alloc((size_t)M * Dff * 2);
    float* ffn_o           = (float*)alloc((size_t)M * Dm * 4);

    const int nElem = M * Dm;
    prep_kernel<<<(nElem + 255) / 256, 256, 0, stream>>>(query, qpos, value,
                                                         q_bf, v_bf, nElem);

    auto tr = [&](const float* w, unsigned short* wt, int K, int N) {
        transpose_w_kernel<<<(K * N + 255) / 256, 256, 0, stream>>>(w, wt, K, N);
    };
    tr(W_v,   Wt_v,   Dm, Dm);
    tr(W_off, Wt_off, Dm, Dm);
    tr(W_att, Wt_att, Dm, 128);
    tr(W_out, Wt_out, Dm, Dm);
    tr(W1,    Wt_1,   Dm, Dff);
    tr(W2,    Wt_2,   Dff, Dm);

    const int gm = (M + 127) / 128;
    // value projection
    gemm_bf16_wmma_kernel<false, false><<<dim3(gm, Dm / 64), 256, 0, stream>>>(
        v_bf, Wt_v, b_v, v_proj, nullptr, M, Dm, Dm);
    // sampling offsets
    gemm_bf16_wmma_kernel<false, false><<<dim3(gm, Dm / 64), 256, 0, stream>>>(
        q_bf, Wt_off, b_off, off_buf, nullptr, M, Dm, Dm);
    // attention logits
    gemm_bf16_wmma_kernel<false, false><<<dim3(gm, 128 / 64), 256, 0, stream>>>(
        q_bf, Wt_att, b_att, attn_buf, nullptr, M, 128, Dm);

    const int groups = M * N_HEADS;
    softmax16_kernel<<<(groups + 255) / 256, 256, 0, stream>>>(attn_buf, groups);

    msda_sample_kernel<<<(groups + 7) / 8, 256, 0, stream>>>(
        v_proj, off_buf, attn_buf, refp, samp, Bn, LQ);

    // output projection
    gemm_bf16_wmma_kernel<false, false><<<dim3(gm, Dm / 64), 256, 0, stream>>>(
        samp, Wt_out, b_out, src2, nullptr, M, Dm, Dm);

    // LN1: q1 = LN(query + src2), also bf16 copy for FFN
    add_ln_kernel<true><<<(M + 7) / 8, 256, 0, stream>>>(
        query, src2, ln1g, ln1b, q1, q1_bf, M);

    // FFN1: relu(q1 @ W1 + b1) -> bf16
    gemm_bf16_wmma_kernel<true, true><<<dim3(gm, Dff / 64), 256, 0, stream>>>(
        q1_bf, Wt_1, b1, nullptr, ffn_h, M, Dff, Dm);
    // FFN2: hidden @ W2 + b2 -> f32
    gemm_bf16_wmma_kernel<false, false><<<dim3(gm, Dm / 64), 256, 0, stream>>>(
        ffn_h, Wt_2, b2, ffn_o, nullptr, M, Dm, Dff);

    // LN2: out = LN(q1 + ffn)
    add_ln_kernel<false><<<(M + 7) / 8, 256, 0, stream>>>(
        q1, ffn_o, ln2g, ln2b, out, nullptr, M);
}
